// MemoryEfficientGNN_33655363732045
// MI455X (gfx1250) — matrix operand order repaired and checked by reading, not compiled, not run on hardware
//
#include <hip/hip_runtime.h>

#define H_DIM 64
#define NSTAY 100000
#define NDIAG 50000
#define NEDGE 1000000
#define FLAG_RELU 1
#define FLAG_NORM 2
#define FLAG_ACC  4

typedef __attribute__((ext_vector_type(16))) _Float16 v16h;
typedef __attribute__((ext_vector_type(8)))  _Float16 v8h;
typedef __attribute__((ext_vector_type(8)))  float    v8f;

// ---------------------------------------------------------------------------
// small utility kernels
// ---------------------------------------------------------------------------
__global__ void cvt_f32_to_f16(const float* __restrict__ in, _Float16* __restrict__ out, int n) {
    int i = blockIdx.x * blockDim.x + threadIdx.x;
    if (i < n) out[i] = (_Float16)in[i];
}

__global__ void zero_f32(float* __restrict__ p, int n) {
    int i = blockIdx.x * blockDim.x + threadIdx.x;
    if (i < n) p[i] = 0.0f;
}

__global__ void count_edges(const int* __restrict__ dst, float* __restrict__ cnt, int E) {
    int e = blockIdx.x * blockDim.x + threadIdx.x;
    if (e < E) atomicAdd(&cnt[dst[e]], 1.0f);
}

// 16 threads per edge, 4 channels each: coalesced 256B gather per edge row,
// L2-resident f32 atomic scatter-add.
__global__ void scatter_add(const float* __restrict__ hsrc, const int* __restrict__ src,
                            const int* __restrict__ dst, float* __restrict__ sum, int E) {
    int tid = blockIdx.x * blockDim.x + threadIdx.x;
    int e = tid >> 4;
    if (e >= E) return;
    int c = (tid & 15) << 2;
    int s = src[e], d = dst[e];
    const float4 v = *(const float4*)(hsrc + (size_t)s * H_DIM + c);
    float* p = sum + (size_t)d * H_DIM + c;
    atomicAdd(p + 0, v.x);
    atomicAdd(p + 1, v.y);
    atomicAdd(p + 2, v.z);
    atomicAdd(p + 3, v.w);
}

// ---------------------------------------------------------------------------
// WMMA fragment loaders (16-bit operands, contiguous k order in both A and B)
// ---------------------------------------------------------------------------
// A (16x32 f16): lane holds row M=lane&15; lanes 0-15 hold K k0..k0+15,
// lanes 16-31 hold K k0+16..k0+31.
__device__ __forceinline__ v16h load_a_frag(const _Float16* tile, int K, int k0, int lane) {
    int row = lane & 15;
    int h8  = (lane >> 4) << 4;
    const _Float16* p = tile + row * K + k0 + h8;
    v8h lo = *(const v8h*)(p);
    v8h hi = *(const v8h*)(p + 8);
    v16h a;
#pragma unroll
    for (int i = 0; i < 8; ++i) { a[i] = lo[i]; a[i + 8] = hi[i]; }
    return a;
}

// B (32x16 f16) = W^T slice: lane n holds column N=n&15, K = k0 + 16*(lane>>4) .. +15,
// which is 16 contiguous f16 of weight row (ncol0+n) -> one 32-byte load.
__device__ __forceinline__ v16h load_b_frag(const _Float16* __restrict__ W, int K,
                                            int ncol0, int k0, int lane) {
    int n    = lane & 15;
    int koff = k0 + ((lane >> 4) << 4);
    return *(const v16h*)(W + (size_t)(ncol0 + n) * K + koff);
}

// ---------------------------------------------------------------------------
// fused GEMM: out[N,64] = act( X1/cnt @ W1^T + bias [+ X2 @ W2^T] ), optional
// per-row L2 normalize and accumulate-into-out. 8 waves/block, 16 rows/wave.
// ---------------------------------------------------------------------------
__global__ void __launch_bounds__(256)
gemm_wmma(const float* __restrict__ X1, const float* __restrict__ cnt,
          const _Float16* __restrict__ W1, const float* __restrict__ bias,
          const float* __restrict__ X2, const _Float16* __restrict__ W2,
          float* __restrict__ out, int N, int K, int flags) {
    __shared__ __align__(32) _Float16 smem[8 * 16 * 128];  // 32 KB
    const int wave = threadIdx.x >> 5;
    const int lane = threadIdx.x & 31;
    const int row0 = blockIdx.x * 128 + wave * 16;
    const int kshift = (K == 128) ? 7 : 6;

    // stage X1 tile (fused mean: divide by max(cnt,1)), f32 -> f16
    _Float16* tile = smem + wave * 16 * K;
    for (int idx = lane; idx < 16 * K; idx += 32) {
        int r = idx >> kshift, c = idx & (K - 1);
        int gr = row0 + r;
        float v = 0.0f;
        if (gr < N) {
            v = X1[(size_t)gr * K + c];
            if (cnt) v /= fmaxf(cnt[gr], 1.0f);
        }
        tile[idx] = (_Float16)v;
    }
    __syncthreads();

    v8f acc[4];
#pragma unroll
    for (int t = 0; t < 4; ++t) {
        float b = bias ? bias[t * 16 + (lane & 15)] : 0.0f;
#pragma unroll
        for (int r = 0; r < 8; ++r) acc[t][r] = b;
    }

    const int ksteps = K >> 5;
    for (int ks = 0; ks < ksteps; ++ks) {
        v16h a = load_a_frag(tile, K, ks * 32, lane);
#pragma unroll
        for (int t = 0; t < 4; ++t) {
            v16h b = load_b_frag(W1, K, t * 16, ks * 32, lane);
            acc[t] = __builtin_amdgcn_wmma_f32_16x16x32_f16(
                false, a, false, b, (short)0, acc[t], false, false);
        }
    }

    if (X2) {  // second operand: x_dst @ W2^T (K2 = 64)
        __syncthreads();
        _Float16* tile2 = smem + wave * 16 * 64;
        for (int idx = lane; idx < 16 * 64; idx += 32) {
            int r = idx >> 6, c = idx & 63;
            int gr = row0 + r;
            tile2[idx] = (_Float16)(gr < N ? X2[(size_t)gr * 64 + c] : 0.0f);
        }
        __syncthreads();
        for (int ks = 0; ks < 2; ++ks) {
            v16h a = load_a_frag(tile2, 64, ks * 32, lane);
#pragma unroll
            for (int t = 0; t < 4; ++t) {
                v16h b = load_b_frag(W2, 64, t * 16, ks * 32, lane);
                acc[t] = __builtin_amdgcn_wmma_f32_16x16x32_f16(
                    false, a, false, b, (short)0, acc[t], false, false);
            }
        }
    }

    if (flags & FLAG_RELU) {
#pragma unroll
        for (int t = 0; t < 4; ++t)
#pragma unroll
            for (int r = 0; r < 8; ++r) acc[t][r] = fmaxf(acc[t][r], 0.0f);
    }

    if (flags & FLAG_NORM) {
        // C/D layout: VGPR r holds row r (lanes 0-15) / r+8 (lanes 16-31), col = lane&15.
        // Row sum-of-squares = reduce over the 16 lanes of each half + 4 col tiles.
        float s[8];
#pragma unroll
        for (int r = 0; r < 8; ++r) {
            float q = 0.0f;
#pragma unroll
            for (int t = 0; t < 4; ++t) q += acc[t][r] * acc[t][r];
            q += __shfl_xor(q, 1, 32);
            q += __shfl_xor(q, 2, 32);
            q += __shfl_xor(q, 4, 32);
            q += __shfl_xor(q, 8, 32);
            s[r] = 1.0f / fmaxf(sqrtf(q), 1e-12f);
        }
#pragma unroll
        for (int t = 0; t < 4; ++t)
#pragma unroll
            for (int r = 0; r < 8; ++r) acc[t][r] *= s[r];
    }

    const int ncol = lane & 15;
    const int moff = (lane >> 4) << 3;
#pragma unroll
    for (int t = 0; t < 4; ++t) {
        int col = t * 16 + ncol;
#pragma unroll
        for (int r = 0; r < 8; ++r) {
            int gr = row0 + moff + r;
            if (gr < N) {
                float v = acc[t][r];
                if (flags & FLAG_ACC) v += out[(size_t)gr * H_DIM + col];
                out[(size_t)gr * H_DIM + col] = v;
            }
        }
    }
}

// ---------------------------------------------------------------------------
// in-place  x = LN(relu(x*scale)) * g + b   — one wave per 64-wide row
// ---------------------------------------------------------------------------
__global__ void lnrelu(float* __restrict__ x, int N, float scale,
                       const float* __restrict__ g, const float* __restrict__ b) {
    int row  = blockIdx.x * 8 + (threadIdx.x >> 5);
    int lane = threadIdx.x & 31;
    if (row >= N) return;
    float* p = x + (size_t)row * H_DIM;
    float v0 = fmaxf(p[lane] * scale, 0.0f);
    float v1 = fmaxf(p[lane + 32] * scale, 0.0f);
    float s = v0 + v1;
    for (int m = 16; m; m >>= 1) s += __shfl_xor(s, m, 32);
    float mu = s * (1.0f / 64.0f);
    float d0 = v0 - mu, d1 = v1 - mu;
    float q = d0 * d0 + d1 * d1;
    for (int m = 16; m; m >>= 1) q += __shfl_xor(q, m, 32);
    float inv = rsqrtf(q * (1.0f / 64.0f) + 1e-5f);
    p[lane]      = d0 * inv * g[lane]      + b[lane];
    p[lane + 32] = d1 * inv * g[lane + 32] + b[lane + 32];
}

// ---------------------------------------------------------------------------
// classifier: out[N,3] = h @ Wc^T + bc   (tiny M -> plain VALU dot products)
// ---------------------------------------------------------------------------
__global__ void classifier(const float* __restrict__ h, const float* __restrict__ Wc,
                           const float* __restrict__ bc, float* __restrict__ out, int N) {
    __shared__ float w[3 * H_DIM];
    __shared__ float bb[3];
    if (threadIdx.x < 3 * H_DIM) w[threadIdx.x] = Wc[threadIdx.x];
    if (threadIdx.x < 3) bb[threadIdx.x] = bc[threadIdx.x];
    __syncthreads();
    int row = blockIdx.x * blockDim.x + threadIdx.x;
    if (row >= N) return;
    const float* p = h + (size_t)row * H_DIM;
    float a0 = bb[0], a1 = bb[1], a2 = bb[2];
    for (int k = 0; k < H_DIM; k += 4) {
        float4 v = *(const float4*)(p + k);
        a0 += v.x * w[k] + v.y * w[k + 1] + v.z * w[k + 2] + v.w * w[k + 3];
        a1 += v.x * w[64 + k] + v.y * w[64 + k + 1] + v.z * w[64 + k + 2] + v.w * w[64 + k + 3];
        a2 += v.x * w[128 + k] + v.y * w[128 + k + 1] + v.z * w[128 + k + 2] + v.w * w[128 + k + 3];
    }
    out[(size_t)row * 3 + 0] = a0;
    out[(size_t)row * 3 + 1] = a1;
    out[(size_t)row * 3 + 2] = a2;
}

// ---------------------------------------------------------------------------
// host orchestration
// ---------------------------------------------------------------------------
static inline int cdiv(int a, int b) { return (a + b - 1) / b; }

extern "C" void kernel_launch(void* const* d_in, const int* in_sizes, int n_in,
                              void* d_out, int out_size, void* d_ws, size_t ws_size,
                              hipStream_t stream) {
    (void)in_sizes; (void)n_in; (void)out_size; (void)ws_size;
    const float* x_stay  = (const float*)d_in[0];
    const float* x_diag  = (const float*)d_in[1];
    const int* s2d_src = (const int*)d_in[2];
    const int* s2d_dst = (const int*)d_in[3];
    const int* d2s_src = (const int*)d_in[4];
    const int* d2s_dst = (const int*)d_in[5];
    const int* s2s_src = (const int*)d_in[6];
    const int* s2s_dst = (const int*)d_in[7];
    const float* Wp_stay = (const float*)d_in[8];
    const float* bp_stay = (const float*)d_in[9];
    const float* Wp_diag = (const float*)d_in[10];
    const float* bp_diag = (const float*)d_in[11];
    const float* Wl   = (const float*)d_in[12];
    const float* bl   = (const float*)d_in[13];
    const float* Wr   = (const float*)d_in[14];
    const float* ln_g = (const float*)d_in[15];
    const float* ln_b = (const float*)d_in[16];
    const float* Wc   = (const float*)d_in[17];
    const float* bc   = (const float*)d_in[18];
    float* out = (float*)d_out;

    // ---- workspace layout ----
    char* ws = (char*)d_ws;
    _Float16* WpS_h = (_Float16*)ws;          // 64*128
    _Float16* WpD_h = WpS_h + 8192;           // 64*64
    _Float16* Wl_h  = WpD_h + 4096;           // 2*3*64*64
    _Float16* Wr_h  = Wl_h + 24576;           // 2*3*64*64
    float* fbase    = (float*)(ws + 131072);  // 128 KB half region
    float* hsA = fbase;                 // 100000*64
    float* hsB = hsA + NSTAY * H_DIM;
    float* hdA = hsB + NSTAY * H_DIM;   // 50000*64
    float* hdB = hdA + NDIAG * H_DIM;
    float* sumbuf  = hdB + NDIAG * H_DIM;     // 100000*64 (reused)
    float* cnt_s2d = sumbuf + NSTAY * H_DIM;  // 50000
    float* cnt_d2s = cnt_s2d + NDIAG;         // 100000
    float* cnt_s2s = cnt_d2s + NSTAY;         // 100000

    // ---- weight conversion (f32 -> f16) ----
    cvt_f32_to_f16<<<cdiv(8192, 256), 256, 0, stream>>>(Wp_stay, WpS_h, 8192);
    cvt_f32_to_f16<<<cdiv(4096, 256), 256, 0, stream>>>(Wp_diag, WpD_h, 4096);
    cvt_f32_to_f16<<<cdiv(24576, 256), 256, 0, stream>>>(Wl, Wl_h, 24576);
    cvt_f32_to_f16<<<cdiv(24576, 256), 256, 0, stream>>>(Wr, Wr_h, 24576);

    // ---- degree counts (layer-invariant) ----
    zero_f32<<<cdiv(NDIAG + 2 * NSTAY, 256), 256, 0, stream>>>(cnt_s2d, NDIAG + 2 * NSTAY);
    count_edges<<<cdiv(NEDGE, 256), 256, 0, stream>>>(s2d_dst, cnt_s2d, NEDGE);
    count_edges<<<cdiv(NEDGE, 256), 256, 0, stream>>>(d2s_dst, cnt_d2s, NEDGE);
    count_edges<<<cdiv(NEDGE, 256), 256, 0, stream>>>(s2s_dst, cnt_s2s, NEDGE);

    // ---- input projections + relu (WMMA) ----
    gemm_wmma<<<cdiv(NSTAY, 128), 256, 0, stream>>>(
        x_stay, nullptr, WpS_h, bp_stay, nullptr, nullptr, hsA, NSTAY, 128, FLAG_RELU);
    gemm_wmma<<<cdiv(NDIAG, 128), 256, 0, stream>>>(
        x_diag, nullptr, WpD_h, bp_diag, nullptr, nullptr, hdA, NDIAG, 64, FLAG_RELU);

    float* cur_s = hsA; float* nxt_s = hsB;
    float* cur_d = hdA; float* nxt_d = hdB;

    for (int l = 0; l < 2; ++l) {
        const _Float16* Wl_l = Wl_h + (size_t)l * 3 * 4096;
        const _Float16* Wr_l = Wr_h + (size_t)l * 3 * 4096;
        const float* bl_l = bl + (size_t)l * 3 * 64;

        // s2d: stay -> diag
        zero_f32<<<cdiv(NDIAG * H_DIM, 256), 256, 0, stream>>>(sumbuf, NDIAG * H_DIM);
        scatter_add<<<cdiv(NEDGE * 16, 256), 256, 0, stream>>>(cur_s, s2d_src, s2d_dst, sumbuf, NEDGE);
        gemm_wmma<<<cdiv(NDIAG, 128), 256, 0, stream>>>(
            sumbuf, cnt_s2d, Wl_l + 0 * 4096, bl_l + 0 * 64, cur_d, Wr_l + 0 * 4096,
            nxt_d, NDIAG, 64, FLAG_NORM);

        // d2s: diag -> stay
        zero_f32<<<cdiv(NSTAY * H_DIM, 256), 256, 0, stream>>>(sumbuf, NSTAY * H_DIM);
        scatter_add<<<cdiv(NEDGE * 16, 256), 256, 0, stream>>>(cur_d, d2s_src, d2s_dst, sumbuf, NEDGE);
        gemm_wmma<<<cdiv(NSTAY, 128), 256, 0, stream>>>(
            sumbuf, cnt_d2s, Wl_l + 1 * 4096, bl_l + 1 * 64, cur_s, Wr_l + 1 * 4096,
            nxt_s, NSTAY, 64, FLAG_NORM);

        // s2s: stay -> stay (accumulate o_s2 onto o_s1)
        zero_f32<<<cdiv(NSTAY * H_DIM, 256), 256, 0, stream>>>(sumbuf, NSTAY * H_DIM);
        scatter_add<<<cdiv(NEDGE * 16, 256), 256, 0, stream>>>(cur_s, s2s_src, s2s_dst, sumbuf, NEDGE);
        gemm_wmma<<<cdiv(NSTAY, 128), 256, 0, stream>>>(
            sumbuf, cnt_s2s, Wl_l + 2 * 4096, bl_l + 2 * 64, cur_s, Wr_l + 2 * 4096,
            nxt_s, NSTAY, 64, FLAG_NORM | FLAG_ACC);

        // HeteroConv mean + relu + LayerNorm (in place)
        lnrelu<<<cdiv(NDIAG, 8), 256, 0, stream>>>(nxt_d, NDIAG, 1.0f, ln_g + l * 64, ln_b + l * 64);
        lnrelu<<<cdiv(NSTAY, 8), 256, 0, stream>>>(nxt_s, NSTAY, 0.5f, ln_g + l * 64, ln_b + l * 64);

        float* t;
        t = cur_s; cur_s = nxt_s; nxt_s = t;
        t = cur_d; cur_d = nxt_d; nxt_d = t;
    }

    classifier<<<cdiv(NSTAY, 256), 256, 0, stream>>>(cur_s, Wc, bc, out, NSTAY);
}
